// GNNCriticEncoder_26637387170385
// MI455X (gfx1250) — compile-verified, matrix operand order checked by reading, not run on hardware
//
#include <hip/hip_runtime.h>
#include <hip/hip_bf16.h>

typedef __attribute__((ext_vector_type(16))) _Float16 v16h;
typedef __attribute__((ext_vector_type(8)))  float    v8f;
typedef __attribute__((ext_vector_type(4)))  unsigned u32x4;
typedef __attribute__((ext_vector_type(2)))  unsigned u32x2;
typedef __attribute__((ext_vector_type(4)))  float    f32x4;

#define OBS_LEN 1056
#define BPB 8

// ---- LDS layout (bytes) ----
#define OFF_WVT    0        // [128][64]  f16 (Wv^T, K padded 40->64)
#define OFF_WPT    16384    // [128][32]  f16 (Wp^T, K padded 24->32)
#define OFF_W0T    24576    // [128][128] f16 (w0^T)
#define OFF_W1T    57344    // [128][128] f16 (w1^T)
#define OFF_BIAS   90112    // [256] f32 (bv | bp)
#define OFF_AVEC   91136    // [2][2][4][32] f32 (layer, src/dst, head, d)
#define OFF_VEHA   93184    // [16][64] f16 (veh rows, K zero-padded 40->64)
#define OFF_PEDA   95232    // [16][32] f16 (ped rows, K zero-padded 24->32)
#define OFF_XF16   96256    // [32][128] f16
#define OFF_HF32   104448   // [32][128] f32
#define OFF_HT16   120832   // [128][32] f16 (h transposed)
#define OFF_ATTN   129024   // [4][32][32] f16
#define OFF_AS     137216   // [4][32] f32 (alpha_src, head-major)
#define OFF_AD     137728   // [4][32] f32 (alpha_dst, head-major)
#define OFF_ALIVE  138240   // [32] f32
#define OFF_SCAL   138368   // [4] f32
#define OFF_XF32   138384   // [32][128] f32
#define LDS_BYTES  154768

__device__ __forceinline__ v8f wmma_f16(v16h a, v16h b, v8f c) {
  return __builtin_amdgcn_wmma_f32_16x16x32_f16(
      /*neg_a=*/false, a, /*neg_b=*/false, b,
      /*c_mod=*/(short)0, c, /*reuse_a=*/false, /*reuse_b=*/false);
}

// 16x32 f16 fragment (A layout; B uses A-layout of B^T) = two b128 LDS loads.
__device__ __forceinline__ v16h load_frag(const _Float16* base, int row,
                                          int stride, int kbase, int lane) {
  union { v16h h; u32x4 q[2]; } r;
  const _Float16* p = base + row * stride + kbase + ((lane >> 4) << 3);
  r.q[0] = *(const u32x4*)(p);
  r.q[1] = *(const u32x4*)(p + 16);
  return r.h;
}

// Pack v8f accumulator into 8 f16 halfs (one 16B LDS store).
__device__ __forceinline__ u32x4 pack_f16x8(v8f a) {
  union { _Float16 h[8]; u32x4 q; } pk;
  #pragma unroll
  for (int v = 0; v < 8; ++v) pk.h[v] = (_Float16)a[v];
  return pk.q;
}

// ---- Pre-pass: pack transposed f16 weights into workspace ----
// Layout (halfs): WvT[128][64] | WpT[128][32] | w0T[128][128] | w1T[128][128]
__global__ __launch_bounds__(256) void gnn_pack(
    const float* __restrict__ Wv, const float* __restrict__ Wp,
    const float* __restrict__ w0, const float* __restrict__ w1,
    _Float16* __restrict__ wpack) {
  int i = blockIdx.x * 256 + threadIdx.x;
  if (i >= 45056) return;
  float val;
  if (i < 8192)        { int n = i >> 6, k = i & 63;              val = (k < 40) ? Wv[k * 128 + n] : 0.0f; }
  else if (i < 12288)  { int j = i - 8192;  int n = j >> 5, k = j & 31;  val = (k < 24) ? Wp[k * 128 + n] : 0.0f; }
  else if (i < 28672)  { int j = i - 12288; int n = j >> 7, k = j & 127; val = w0[k * 128 + n]; }
  else                 { int j = i - 28672; int n = j >> 7, k = j & 127; val = w1[k * 128 + n]; }
  wpack[i] = (_Float16)val;
}

// ---- Main kernel: one block handles BPB batch elements ----
__global__ __launch_bounds__(256) void gnn_main(
    const float* __restrict__ gobs,
    const float* __restrict__ bv, const float* __restrict__ bp,
    const float* __restrict__ asrc0, const float* __restrict__ adst0,
    const float* __restrict__ asrc1, const float* __restrict__ adst1,
    const _Float16* __restrict__ wpack, float* __restrict__ out, int Bsz) {
  extern __shared__ char smem[];
  _Float16* wvT   = (_Float16*)(smem + OFF_WVT);
  _Float16* wpT   = (_Float16*)(smem + OFF_WPT);
  _Float16* w0T   = (_Float16*)(smem + OFF_W0T);
  _Float16* w1T   = (_Float16*)(smem + OFF_W1T);
  float*    biasVP= (float*)(smem + OFF_BIAS);
  float*    avec  = (float*)(smem + OFF_AVEC);
  _Float16* vehA  = (_Float16*)(smem + OFF_VEHA);
  _Float16* pedA  = (_Float16*)(smem + OFF_PEDA);
  _Float16* xF16  = (_Float16*)(smem + OFF_XF16);
  float*    hF32  = (float*)(smem + OFF_HF32);
  _Float16* hT16  = (_Float16*)(smem + OFF_HT16);
  _Float16* attn16= (_Float16*)(smem + OFF_ATTN);
  float*    alphaS= (float*)(smem + OFF_AS);
  float*    alphaD= (float*)(smem + OFF_AD);
  float*    aliveE= (float*)(smem + OFF_ALIVE);
  float*    scal  = (float*)(smem + OFF_SCAL);
  float*    xF32  = (float*)(smem + OFF_XF32);

  const int t = threadIdx.x;
  const int lane = t & 31;
  const int wave = t >> 5;
  const int l16 = lane & 15;
  const int hiM = (lane >> 4) << 3;

  // One-time weight staging into LDS (f16 weights contiguous in wpack)
  {
    const unsigned* src = (const unsigned*)wpack;
    unsigned* dst = (unsigned*)wvT;   // wvT|wpT|w0T|w1T contiguous
    for (int i = t; i < 22528; i += 256) dst[i] = src[i];
    for (int i = t; i < 128; i += 256) {
      biasVP[i] = bv[i]; biasVP[128 + i] = bp[i];
      avec[i] = asrc0[i]; avec[128 + i] = adst0[i];
      avec[256 + i] = asrc1[i]; avec[384 + i] = adst1[i];
    }
  }

  for (int bi = 0; bi < BPB; ++bi) {
    int b = blockIdx.x * BPB + bi;
    if (b >= Bsz) break;
    const float* go = gobs + (size_t)b * OBS_LEN;

    __syncthreads();  // protect LDS reuse from previous iteration readers

    // Stage zero-padded f16 A-matrices + alive flags straight from global.
    {
      union { _Float16 h[4]; u32x2 q; } vp;       // 4 veh halfs per thread
      int i0 = t * 4;
      #pragma unroll
      for (int jj = 0; jj < 4; ++jj) {
        int i = i0 + jj;
        int m = i >> 6, k = i & 63;
        vp.h[jj] = (_Float16)((k < 40) ? go[m * 40 + k] : 0.0f);
      }
      *(u32x2*)(vehA + i0) = vp.q;

      union { _Float16 h[2]; unsigned u; } pp;    // 2 ped halfs per thread
      int j0 = t * 2;
      #pragma unroll
      for (int jj = 0; jj < 2; ++jj) {
        int i = j0 + jj;
        int m = i >> 5, k = i & 31;
        pp.h[jj] = (_Float16)((k < 24) ? go[640 + m * 24 + k] : 0.0f);
      }
      *(unsigned*)(pedA + j0) = pp.u;
    }
    if (t < 32) aliveE[t] = (go[1024 + t] >= 0.5f) ? 1.0f : 0.0f;
    if (bi + 1 < BPB && b + 1 < Bsz) {
      const char* nx = (const char*)(go + OBS_LEN);
      if (t >= 128 && t < 162) __builtin_prefetch(nx + (t - 128) * 128, 0, 1);
    }
    __syncthreads();

    if (t == 0) {
      float s = 0.0f;
      for (int i = 0; i < 32; ++i) s += aliveE[i];
      if (s < 0.5f) { for (int i = 0; i < 32; ++i) aliveE[i] = 1.0f; s = 32.0f; }
      scal[0] = __builtin_amdgcn_rcpf(fmaxf(s, 1.0f));
    }

    // ---- Token embedding: xF16[32][128] = [veh@Wv+bv ; ped@Wp+bp] ----
    {
      int col = wave * 16 + l16;               // 8 waves cover N=128
      v8f acc;
      #pragma unroll
      for (int v = 0; v < 8; ++v) acc[v] = biasVP[col];
      #pragma unroll
      for (int c = 0; c < 2; ++c) {            // veh K=40 padded to 64
        v16h bf = load_frag(wvT, col, 64, c * 32, lane);
        v16h af = load_frag(vehA, l16, 64, c * 32, lane);
        acc = wmma_f16(af, bf, acc);
      }
      #pragma unroll
      for (int v = 0; v < 8; ++v)
        xF16[(v + hiM) * 128 + col] = (_Float16)acc[v];

      v8f accp;
      #pragma unroll
      for (int v = 0; v < 8; ++v) accp[v] = biasVP[128 + col];
      {                                        // ped K=24 padded to 32
        v16h bf = load_frag(wpT, col, 32, 0, lane);
        v16h af = load_frag(pedA, l16, 32, 0, lane);
        accp = wmma_f16(af, bf, accp);
      }
      #pragma unroll
      for (int v = 0; v < 8; ++v)
        xF16[(16 + v + hiM) * 128 + col] = (_Float16)accp[v];
    }
    __syncthreads();

    // ---- Two GAT layers ----
    #pragma unroll
    for (int layer = 0; layer < 2; ++layer) {
      const _Float16* wT = layer ? w1T : w0T;
      const float* av = avec + layer * 256;

      // h = x @ w : M=32 (2 tiles), N=128 (8 waves), K=128 (4 chunks)
      {
        int n = wave * 16 + l16;
        v8f acc0, acc1;
        #pragma unroll
        for (int v = 0; v < 8; ++v) { acc0[v] = 0.0f; acc1[v] = 0.0f; }
        #pragma unroll
        for (int c = 0; c < 4; ++c) {
          v16h bf  = load_frag(wT, n, 128, c * 32, lane);
          v16h af0 = load_frag(xF16, l16, 128, c * 32, lane);
          acc0 = wmma_f16(af0, bf, acc0);
          v16h af1 = load_frag(xF16, 16 + l16, 128, c * 32, lane);
          acc1 = wmma_f16(af1, bf, acc1);
        }
        #pragma unroll
        for (int v = 0; v < 8; ++v) {
          int m = v + hiM;
          hF32[m * 128 + n] = acc0[v];
          hF32[(16 + m) * 128 + n] = acc1[v];
        }
        // hT16 rows: 8 consecutive halfs per accumulator -> one b128 each
        *(u32x4*)(hT16 + n * 32 + hiM)      = pack_f16x8(acc0);
        *(u32x4*)(hT16 + n * 32 + 16 + hiM) = pack_f16x8(acc1);
      }
      __syncthreads();

      // alpha_src/dst (head-major): 128 threads, one (token, head) each
      if (t < 128) {
        int n = t >> 2, hh = t & 3;
        float s = 0.0f, d = 0.0f;
        const float* hrow = hF32 + n * 128 + hh * 32;
        const float* a1 = av + hh * 32;
        const float* a2 = av + 128 + hh * 32;
        #pragma unroll
        for (int k4 = 0; k4 < 8; ++k4) {
          f32x4 hv = *(const f32x4*)(hrow + k4 * 4);
          f32x4 x1 = *(const f32x4*)(a1 + k4 * 4);
          f32x4 x2 = *(const f32x4*)(a2 + k4 * 4);
          #pragma unroll
          for (int c = 0; c < 4; ++c) { s += hv[c] * x1[c]; d += hv[c] * x2[c]; }
        }
        alphaS[hh * 32 + n] = s; alphaD[hh * 32 + n] = d;
      }
      __syncthreads();

      // branchless leaky-relu + mask + softmax -> attn16[h][i][j]
      if (t < 128) {
        int i = t >> 2, hh = t & 3;
        float asi = alphaS[hh * 32 + i];
        float ad[32], alv[32];
        #pragma unroll
        for (int j4 = 0; j4 < 8; ++j4) {   // vector LDS reads (b128)
          *(f32x4*)(ad + 4 * j4)  = *(const f32x4*)(alphaD + hh * 32 + 4 * j4);
          *(f32x4*)(alv + 4 * j4) = *(const f32x4*)(aliveE + 4 * j4);
        }
        float p[32];
        float mx = -1e30f;
        #pragma unroll
        for (int j = 0; j < 32; ++j) {
          float e = asi + ad[j];
          e = (e > 0.0f) ? e : 0.2f * e;
          e = (alv[j] > 0.5f) ? e : -1e30f;
          p[j] = e;
          mx = fmaxf(mx, e);
        }
        float sum = 0.0f;
        #pragma unroll
        for (int j = 0; j < 32; ++j) {
          float ex = __expf(p[j] - mx);
          ex = (alv[j] > 0.5f) ? ex : 0.0f;
          p[j] = ex;
          sum += ex;
        }
        float inv = __builtin_amdgcn_rcpf(sum);
        union { _Float16 h[32]; u32x4 q[4]; } ow;
        #pragma unroll
        for (int j = 0; j < 32; ++j) ow.h[j] = (_Float16)(p[j] * inv);
        u32x4* dq = (u32x4*)(attn16 + hh * 1024 + i * 32);
        #pragma unroll
        for (int q4 = 0; q4 < 4; ++q4) dq[q4] = ow.q[q4];
      }
      __syncthreads();

      // out = attn @ h per head (K=32), then ELU -> xF16 / xF32
      {
        int hh = wave & 3, mt = wave >> 2;
        v16h af = load_frag(attn16 + hh * 1024, mt * 16 + l16, 32, 0, lane);
        #pragma unroll
        for (int nt = 0; nt < 2; ++nt) {
          v16h bf = load_frag(hT16, hh * 32 + nt * 16 + l16, 32, 0, lane);
          v8f acc;
          #pragma unroll
          for (int v = 0; v < 8; ++v) acc[v] = 0.0f;
          acc = wmma_f16(af, bf, acc);
          int dcol = hh * 32 + nt * 16 + l16;
          #pragma unroll
          for (int v = 0; v < 8; ++v) {
            int i = mt * 16 + v + hiM;
            float x = acc[v];
            float e = (x > 0.0f) ? x : (__expf(x) - 1.0f);
            xF32[i * 128 + dcol] = e;
            xF16[i * 128 + dcol] = (_Float16)e;
          }
        }
      }
      __syncthreads();
    }

    // ---- Masked mean pool -> out[b][0..127] ----
    if (t < 128) {
      float s = 0.0f;
      #pragma unroll
      for (int i = 0; i < 32; ++i) s += xF32[i * 128 + t] * aliveE[i];
      out[(size_t)b * 128 + t] = s * scal[0];
    }
  }
}

extern "C" void kernel_launch(void* const* d_in, const int* in_sizes, int n_in,
                              void* d_out, int out_size, void* d_ws, size_t ws_size,
                              hipStream_t stream) {
  (void)n_in; (void)out_size; (void)ws_size;
  const float* gobs  = (const float*)d_in[0];
  const float* Wv    = (const float*)d_in[1];
  const float* bv    = (const float*)d_in[2];
  const float* Wp    = (const float*)d_in[3];
  const float* bp    = (const float*)d_in[4];
  const float* w0    = (const float*)d_in[5];
  const float* asrc0 = (const float*)d_in[6];
  const float* adst0 = (const float*)d_in[7];
  const float* w1    = (const float*)d_in[8];
  const float* asrc1 = (const float*)d_in[9];
  const float* adst1 = (const float*)d_in[10];
  _Float16* wpack = (_Float16*)d_ws;
  float* outp = (float*)d_out;

  int Bsz = in_sizes[0] / OBS_LEN;
  int grid = (Bsz + BPB - 1) / BPB;

  gnn_pack<<<176, 256, 0, stream>>>(Wv, Wp, w0, w1, wpack);

  hipFuncSetAttribute(reinterpret_cast<const void*>(&gnn_main),
                      hipFuncAttributeMaxDynamicSharedMemorySize, LDS_BYTES);
  gnn_main<<<grid, 256, LDS_BYTES, stream>>>(gobs, bv, bp, asrc0, adst0,
                                             asrc1, adst1, wpack, outp, Bsz);
}
